// MultiHeadSelfAttention_12910671692485
// MI455X (gfx1250) — matrix operand order repaired and checked by reading, not compiled, not run on hardware
//
#include <hip/hip_runtime.h>

#define SEQ    4096
#define DMODEL 1024
#define NQKV   3072
#define HEADS  16
#define HDIM   64

typedef __attribute__((ext_vector_type(16))) __bf16 v16bf;
typedef __attribute__((ext_vector_type(8)))  float  v8f;
typedef __attribute__((ext_vector_type(4)))  unsigned int v4u;
typedef __attribute__((ext_vector_type(8)))  int v8i;
typedef __attribute__((ext_vector_type(4)))  int v4i;

#if defined(__has_builtin)
#if __has_builtin(__builtin_amdgcn_tensor_load_to_lds)
#define HAVE_TDM 1
#endif
#endif
#ifndef HAVE_TDM
#define HAVE_TDM 0
#endif

#if HAVE_TDM
#if __has_include(<hip/amd_detail/amd_gfx1250_TDM.h>)
#define TDM_6ARG 1          // therock/clang-23 flavor
#else
#define TDM_6ARG 0          // ROCm 7.2 / clang-22 flavor (5 args)
#endif
#endif

union FragB {
  v16bf v;
  uint4 q[2];
  unsigned int   d[8];
  unsigned short u[16];
};

__device__ __forceinline__ unsigned short f2bf(float f) {
  unsigned int u = __float_as_uint(f);
  return (unsigned short)((u + 0x7FFFu + ((u >> 16) & 1u)) >> 16);  // RNE
}

__device__ __forceinline__ v8f wmma_bf16(v16bf a, v16bf b, v8f c) {
  // (neg_a, A, neg_b, B, c_mod, C, reuse_a, reuse_b)
  return __builtin_amdgcn_wmma_f32_16x16x32_bf16(false, a, false, b, (short)0, c,
                                                 false, false);
}

// A operand (16x32 bf16) from row-major [M][K]: lane half h ->
//   elems 0..15 = K (16h .. 16h+15)  (contiguous 32B run)
__device__ __forceinline__ v16bf frag_a(const unsigned short* base, int row, int ldm,
                                        int koff, int lane) {
  const unsigned short* p = base + (row + (lane & 15)) * ldm + koff + ((lane >> 4) << 4);
  FragB f;
  f.q[0] = *(const uint4*)(p);
  f.q[1] = *(const uint4*)(p + 8);
  return f.v;
}

// B operand (32x16 bf16) from row-major [N][K]: lane half h ->
//   elems 0..15 = K (16h .. 16h+15)  (contiguous 32B run)
__device__ __forceinline__ v16bf frag_b(const unsigned short* base, int col, int ldm,
                                        int koff, int lane) {
  const unsigned short* p = base + (col + (lane & 15)) * ldm + koff + ((lane >> 4) << 4);
  FragB f;
  f.q[0] = *(const uint4*)(p);
  f.q[1] = *(const uint4*)(p + 8);
  return f.v;
}

// ---------------------------------------------------------------------------
// Tensor Data Mover: 2D bf16 tile (global -> LDS).  D# layout per
// CDNA5 ISA: group0 = {count, lds_addr,
// global_addr, type=2}; group1 = {data_size=2B, tensor_dim0/1,
// tile_dim0/1, tensor_dim0_stride}; groups 2/3 zero (<=2D tensor).
// ---------------------------------------------------------------------------
__device__ __forceinline__ void tdm_load_2d_bf16(unsigned lds_off, const void* gptr,
                                                 unsigned tensor_d0, unsigned tensor_d1,
                                                 unsigned stride0,
                                                 unsigned tile_d0, unsigned tile_d1) {
#if HAVE_TDM
  unsigned long long ga = (unsigned long long)(uintptr_t)gptr;
  v4u g0;
  g0[0] = 1u;                                            // count = 1 valid D#
  g0[1] = lds_off;                                       // LDS byte address
  g0[2] = (unsigned)(ga & 0xFFFFFFFFu);                  // global_addr[31:0]
  g0[3] = (unsigned)((ga >> 32) & 0x01FFFFFFu) | (2u << 30);  // addr[56:32] | type=2
  v8i g1;
  g1[0] = (int)(1u << 16);                               // data_size = 1 (2 bytes)
  g1[1] = (int)((tensor_d0 & 0xFFFFu) << 16);            // tensor_dim0[15:0] @bit48
  g1[2] = (int)(((tensor_d0 >> 16) & 0xFFFFu) |
                ((tensor_d1 & 0xFFFFu) << 16));          // dim0 hi | dim1 lo
  g1[3] = (int)(((tensor_d1 >> 16) & 0xFFFFu) |
                ((tile_d0 & 0xFFFFu) << 16));            // dim1 hi | tile_dim0
  g1[4] = (int)(tile_d1 & 0xFFFFu);                      // tile_dim1 ; tile_dim2=0
  g1[5] = (int)stride0;                                  // tensor_dim0_stride[31:0]
  g1[6] = 0;                                             // stride0 hi | dim1_stride lo
  g1[7] = 0;                                             // dim1_stride hi (unused, 2D)
  v4i gz = {0, 0, 0, 0};
#if TDM_6ARG
  v8i gz8 = {0, 0, 0, 0, 0, 0, 0, 0};
  __builtin_amdgcn_tensor_load_to_lds(g0, g1, gz, gz, gz8, 0);
#else
  __builtin_amdgcn_tensor_load_to_lds(g0, g1, gz, gz, 0);
#endif
#else
  (void)lds_off; (void)gptr; (void)tensor_d0; (void)tensor_d1;
  (void)stride0; (void)tile_d0; (void)tile_d1;
#endif
}

__device__ __forceinline__ void tdm_wait0() {
#if defined(__has_builtin) && __has_builtin(__builtin_amdgcn_s_wait_tensorcnt)
  __builtin_amdgcn_s_wait_tensorcnt(0);
#else
  asm volatile("s_wait_tensorcnt 0x0" ::: "memory");
#endif
}

// ---------------------------------------------------------------------------
// Kernel 1: qkv = x @ w_qkv^T + b_qkv ; scatter Q(/8), K as [h][n][64] bf16,
//           V transposed as [h][64][n] bf16.
// ---------------------------------------------------------------------------
__global__ __launch_bounds__(256)
void qkv_kernel(const float* __restrict__ x, const float* __restrict__ w,
                const float* __restrict__ bias,
                unsigned short* __restrict__ Q, unsigned short* __restrict__ K,
                unsigned short* __restrict__ Vt) {
  __shared__ __align__(16) unsigned short lA[128 * 32];
  __shared__ __align__(16) unsigned short lB[128 * 32];
  const int tid = threadIdx.x, lane = tid & 31, wid = tid >> 5;
  const int waveM = wid & 1, waveN = wid >> 1;          // 2 x 4 waves
  const int m0 = blockIdx.x * 128, n0 = blockIdx.y * 128;

  v8f acc[4][2];
  const v8f vzero = {0.f, 0.f, 0.f, 0.f, 0.f, 0.f, 0.f, 0.f};
  for (int i = 0; i < 4; ++i)
    for (int j = 0; j < 2; ++j) acc[i][j] = vzero;

#pragma unroll 1
  for (int kk = 0; kk < DMODEL; kk += 32) {
    for (int it = 0; it < 4; ++it) {                    // 128x32 f32 -> bf16
      int e = tid + it * 256;                           // 1024 float4 chunks
      int row = e >> 3, c4 = (e & 7) << 2;
      float4 va = *(const float4*)(x + (size_t)(m0 + row) * DMODEL + kk + c4);
      unsigned short* pa = &lA[row * 32 + c4];
      pa[0] = f2bf(va.x); pa[1] = f2bf(va.y); pa[2] = f2bf(va.z); pa[3] = f2bf(va.w);
      float4 vb = *(const float4*)(w + (size_t)(n0 + row) * DMODEL + kk + c4);
      unsigned short* pb = &lB[row * 32 + c4];
      pb[0] = f2bf(vb.x); pb[1] = f2bf(vb.y); pb[2] = f2bf(vb.z); pb[3] = f2bf(vb.w);
    }
    __syncthreads();
    v16bf af[4], bfr[2];
    for (int fm = 0; fm < 4; ++fm) af[fm] = frag_a(lA, waveM * 64 + fm * 16, 32, 0, lane);
    for (int fn = 0; fn < 2; ++fn) bfr[fn] = frag_b(lB, waveN * 32 + fn * 16, 32, 0, lane);
    for (int fm = 0; fm < 4; ++fm)
      for (int fn = 0; fn < 2; ++fn)
        acc[fm][fn] = wmma_bf16(af[fm], bfr[fn], acc[fm][fn]);
    __syncthreads();
  }

  const int half = lane >> 4, col = lane & 15;
  for (int fm = 0; fm < 4; ++fm)
    for (int fn = 0; fn < 2; ++fn) {
      int f = n0 + waveN * 32 + fn * 16 + col;
      float bv = bias[f];
      int c = f >> 10, rem = f & 1023, hh = rem >> 6, dd = rem & 63;
      for (int r = 0; r < 8; ++r) {
        int m = m0 + waveM * 64 + fm * 16 + r + 8 * half;
        float v = acc[fm][fn][r] + bv;
        if (c == 0)      Q[((size_t)hh * SEQ + m) * HDIM + dd] = f2bf(v * 0.125f);
        else if (c == 1) K[((size_t)hh * SEQ + m) * HDIM + dd] = f2bf(v);
        else             Vt[((size_t)hh * HDIM + dd) * SEQ + m] = f2bf(v);
      }
    }
}

// ---------------------------------------------------------------------------
// Kernel 2: flash attention per (head, 128-query tile).  Computes S^T = K.Q^T
// so each lane owns one query; online softmax needs a single shfl_xor(16).
// O^T accumulated via WMMA (V^T x P^T).  K/V tiles staged via the Tensor
// Data Mover (wave 0 issues 2 descriptor loads, s_wait_tensorcnt, barrier).
// ---------------------------------------------------------------------------
__global__ __launch_bounds__(256)
void attn_kernel(const unsigned short* __restrict__ Q,
                 const unsigned short* __restrict__ K,
                 const unsigned short* __restrict__ Vt,
                 unsigned short* __restrict__ ctx) {
  __shared__ __align__(16) unsigned short lK[64 * 64];   // [key][d]
  __shared__ __align__(16) unsigned short lV[64 * 64];   // [d][key]
  const int tid = threadIdx.x, lane = tid & 31, wid = tid >> 5;
  const int half = lane >> 4, qcol = lane & 15;
  const int hh = blockIdx.y;
  const int qw = blockIdx.x * 128 + wid * 16;            // wave's 16 queries
  const unsigned short* Qh = Q + (size_t)hh * SEQ * HDIM;
  const unsigned short* Kh = K + (size_t)hh * SEQ * HDIM;
  const unsigned short* Vh = Vt + (size_t)hh * HDIM * SEQ;
#if HAVE_TDM
  const unsigned ldsK = (unsigned)(uintptr_t)(void*)lK;  // LDS byte offsets
  const unsigned ldsV = (unsigned)(uintptr_t)(void*)lV;
#endif

  // Q as WMMA B-operand (column = query), resident in registers (d = 2x32)
  v16bf bq[2];
  for (int s = 0; s < 2; ++s) {
    const unsigned short* p = Qh + (size_t)(qw + qcol) * HDIM + s * 32 + half * 16;
    FragB fr;
    fr.q[0] = *(const uint4*)(p);
    fr.q[1] = *(const uint4*)(p + 8);
    bq[s] = fr.v;
  }

  float m_run = -1e30f, l_run = 0.f;
  const v8f vzero = {0.f, 0.f, 0.f, 0.f, 0.f, 0.f, 0.f, 0.f};
  v8f o[4];
  for (int i = 0; i < 4; ++i) o[i] = vzero;

#pragma unroll 1
  for (int key0 = 0; key0 < SEQ; key0 += 64) {
#if HAVE_TDM
    if (wid == 0) {
      // K tile: 64 rows (keys) x 64 cols (d), row stride HDIM
      tdm_load_2d_bf16(ldsK, Kh + (size_t)key0 * HDIM, HDIM, SEQ, HDIM, 64, 64);
      // V^T tile: 64 rows (d) x 64 cols (keys), row stride SEQ
      tdm_load_2d_bf16(ldsV, Vh + key0, SEQ, HDIM, SEQ, 64, 64);
      tdm_wait0();
    }
#else
    for (int it = 0; it < 2; ++it) {                     // 2 x 8KB tiles
      int e = tid + it * 256;                            // 512 16B chunks
      int off = e << 3, row = off >> 6, cc = off & 63;
      *(uint4*)&lK[off] = *(const uint4*)(Kh + (size_t)(key0 + row) * HDIM + cc);
      *(uint4*)&lV[off] = *(const uint4*)(Vh + (size_t)row * SEQ + key0 + cc);
    }
#endif
    if (key0 + 64 < SEQ) {
      __builtin_prefetch(Kh + (size_t)(key0 + 64) * HDIM, 0, 1);
      __builtin_prefetch(Vh + key0 + 64, 0, 1);
    }
    __syncthreads();

    // S^T tiles: 4 x (16 keys x 16 queries), inner dim d = 64 (2 WMMA steps)
    v8f s[4];
    for (int kg = 0; kg < 4; ++kg) {
      v8f sa = vzero;
      for (int st = 0; st < 2; ++st)
        sa = wmma_bf16(frag_a(lK, kg * 16, 64, st * 32, lane), bq[st], sa);
      s[kg] = sa;
    }

    // online softmax (per query = per lane%16; halves combined via xor-16)
    float tmax = -1e30f;
    for (int kg = 0; kg < 4; ++kg)
      for (int r = 0; r < 8; ++r) tmax = fmaxf(tmax, s[kg][r]);
    tmax = fmaxf(tmax, __shfl_xor(tmax, 16, 32));
    float m_new = fmaxf(m_run, tmax);
    float corr = __expf(m_run - m_new);
    float lsum = 0.f;
    for (int kg = 0; kg < 4; ++kg)
      for (int r = 0; r < 8; ++r) {
        float p = __expf(s[kg][r] - m_new);
        s[kg][r] = p;
        lsum += p;
      }
    lsum += __shfl_xor(lsum, 16, 32);
    l_run = l_run * corr + lsum;
    m_run = m_new;
    for (int dt = 0; dt < 4; ++dt)
      for (int r = 0; r < 8; ++r) o[dt][r] *= corr;

    // O^T += V^T x P^T.  Build P^T B-fragments in-register: my half supplies
    // elems 8h+r, partner lane (xor 16) supplies elems 8(1-h)+r.
    for (int ks = 0; ks < 2; ++ks) {
      unsigned int mine[4], send[4];
      int kgm = ks * 2 + half, kgs = ks * 2 + (1 - half);
      for (int j = 0; j < 4; ++j) {
        mine[j] = (unsigned)f2bf(s[kgm][2 * j]) | ((unsigned)f2bf(s[kgm][2 * j + 1]) << 16);
        send[j] = (unsigned)f2bf(s[kgs][2 * j]) | ((unsigned)f2bf(s[kgs][2 * j + 1]) << 16);
      }
      FragB pb;
      for (int j = 0; j < 4; ++j) {
        unsigned int rcv = __shfl_xor(send[j], 16, 32);
        pb.d[4 * half + j] = mine[j];
        pb.d[4 * (1 - half) + j] = rcv;
      }
      for (int dt = 0; dt < 4; ++dt)
        o[dt] = wmma_bf16(frag_a(lV, dt * 16, 64, ks * 32, lane), pb.v, o[dt]);
    }
    __syncthreads();
  }

  float inv = 1.0f / l_run;
  for (int dt = 0; dt < 4; ++dt)
    for (int r = 0; r < 8; ++r) {
      int d = dt * 16 + r + 8 * half;
      ctx[(size_t)(qw + qcol) * DMODEL + hh * HDIM + d] = f2bf(o[dt][r] * inv);
    }
}

// ---------------------------------------------------------------------------
// Kernel 3: out = ctx @ w_out^T + b_out   (f32 output)
// ---------------------------------------------------------------------------
__global__ __launch_bounds__(256)
void out_kernel(const unsigned short* __restrict__ ctx, const float* __restrict__ w,
                const float* __restrict__ bias, float* __restrict__ out) {
  __shared__ __align__(16) unsigned short lA[128 * 32];
  __shared__ __align__(16) unsigned short lB[128 * 32];
  const int tid = threadIdx.x, lane = tid & 31, wid = tid >> 5;
  const int waveM = wid & 1, waveN = wid >> 1;
  const int m0 = blockIdx.x * 128, n0 = blockIdx.y * 128;

  v8f acc[4][2];
  const v8f vzero = {0.f, 0.f, 0.f, 0.f, 0.f, 0.f, 0.f, 0.f};
  for (int i = 0; i < 4; ++i)
    for (int j = 0; j < 2; ++j) acc[i][j] = vzero;

#pragma unroll 1
  for (int kk = 0; kk < DMODEL; kk += 32) {
    for (int it = 0; it < 2; ++it) {                     // A tile: bf16 copy
      int e = tid + it * 256;
      int off = e << 3, row = off >> 5, cc = off & 31;
      *(uint4*)&lA[off] = *(const uint4*)(ctx + (size_t)(m0 + row) * DMODEL + kk + cc);
    }
    for (int it = 0; it < 4; ++it) {                     // B tile: f32 -> bf16
      int e = tid + it * 256;
      int row = e >> 3, c4 = (e & 7) << 2;
      float4 vb = *(const float4*)(w + (size_t)(n0 + row) * DMODEL + kk + c4);
      unsigned short* pb = &lB[row * 32 + c4];
      pb[0] = f2bf(vb.x); pb[1] = f2bf(vb.y); pb[2] = f2bf(vb.z); pb[3] = f2bf(vb.w);
    }
    __syncthreads();
    v16bf af[4], bfr[2];
    for (int fm = 0; fm < 4; ++fm) af[fm] = frag_a(lA, waveM * 64 + fm * 16, 32, 0, lane);
    for (int fn = 0; fn < 2; ++fn) bfr[fn] = frag_b(lB, waveN * 32 + fn * 16, 32, 0, lane);
    for (int fm = 0; fm < 4; ++fm)
      for (int fn = 0; fn < 2; ++fn)
        acc[fm][fn] = wmma_bf16(af[fm], bfr[fn], acc[fm][fn]);
    __syncthreads();
  }

  const int half = lane >> 4, col = lane & 15;
  for (int fm = 0; fm < 4; ++fm)
    for (int fn = 0; fn < 2; ++fn) {
      int f = n0 + waveN * 32 + fn * 16 + col;
      float bv = bias[f];
      for (int r = 0; r < 8; ++r) {
        int m = m0 + waveM * 64 + fm * 16 + r + 8 * half;
        out[(size_t)m * DMODEL + f] = acc[fm][fn][r] + bv;
      }
    }
}

extern "C" void kernel_launch(void* const* d_in, const int* in_sizes, int n_in,
                              void* d_out, int out_size, void* d_ws, size_t ws_size,
                              hipStream_t stream) {
  (void)in_sizes; (void)n_in; (void)out_size; (void)ws_size;
  const float* x     = (const float*)d_in[0];
  const float* w_qkv = (const float*)d_in[1];
  const float* b_qkv = (const float*)d_in[2];
  const float* w_out = (const float*)d_in[3];
  const float* b_out = (const float*)d_in[4];
  float* out = (float*)d_out;

  // bf16 workspace: Q | K | Vt (each 16*4096*64) | ctx (4096*1024)  = 32 MB
  unsigned short* Q   = (unsigned short*)d_ws;
  unsigned short* K   = Q + (size_t)HEADS * SEQ * HDIM;
  unsigned short* Vt  = K + (size_t)HEADS * SEQ * HDIM;
  unsigned short* ctx = Vt + (size_t)HEADS * SEQ * HDIM;

  qkv_kernel<<<dim3(SEQ / 128, NQKV / 128), 256, 0, stream>>>(x, w_qkv, b_qkv, Q, K, Vt);
  attn_kernel<<<dim3(SEQ / 128, HEADS), 256, 0, stream>>>(Q, K, Vt, ctx);
  out_kernel<<<dim3(SEQ / 128, DMODEL / 128), 256, 0, stream>>>(ctx, w_out, b_out, out);
}
